// GIN_Backbone_78194174591254
// MI455X (gfx1250) — matrix operand order repaired and checked by reading, not compiled, not run on hardware
//
#include <hip/hip_runtime.h>

#define HID 256
#define N_NODES 100000
#define N_EDGES 300000
#define LN_EPS 1e-5f
#define USE_ASYNC_STAGE 1

typedef __attribute__((ext_vector_type(16))) __bf16 v16bf;
typedef __attribute__((ext_vector_type(8)))  float  v8f;

union BFrag {
    struct { uint4 lo, hi; } q;
    v16bf v;
};

// ---------------------------------------------------------------------------
// Pack fp32 weights into per-lane WMMA B-fragment layout (bf16).
// B fragment (32x16, 16-bit): lane l holds column N = l&15, K = (l>>4)*8 + (j&7) + (j>>3)*16,
// j = 0..15. Packed offset: mat*65536 + (kkIdx*16+n0Idx)*512 + lane*16 + j.
// ---------------------------------------------------------------------------
__global__ __launch_bounds__(256) void pack_weights(const float* __restrict__ W1,
                                                    const float* __restrict__ W2,
                                                    __bf16* __restrict__ Wp) {
    int idx = blockIdx.x * 256 + threadIdx.x;      // 0 .. 393215
    int j     = idx & 15;
    int lane  = (idx >> 4) & 31;
    int frag  = idx >> 9;
    int n0Idx = frag & 15;
    int kkIdx = (frag >> 4) & 7;
    int mat   = frag >> 7;                          // 0..5 = layer*2 + which
    int which = mat & 1;
    int layer = mat >> 1;
    int col = n0Idx * 16 + (lane & 15);
    int K   = kkIdx * 32 + ((lane >> 4) << 3) + (j & 7) + ((j >> 3) << 4);
    const float* W = which ? W2 : W1;
    Wp[idx] = (__bf16)W[layer * 65536 + K * HID + col];
}

// ---------------------------------------------------------------------------
// z = (1 + eps) * h   (float4 per thread)
// ---------------------------------------------------------------------------
__global__ __launch_bounds__(256) void z_init(const float* __restrict__ h,
                                              const float* __restrict__ epsP,
                                              float* __restrict__ z) {
    int i = blockIdx.x * 256 + threadIdx.x;        // 0 .. 6.4M-1
    float e = 1.0f + *epsP;
    float4 v = ((const float4*)h)[i];
    v.x *= e; v.y *= e; v.z *= e; v.w *= e;
    ((float4*)z)[i] = v;
}

// ---------------------------------------------------------------------------
// z[dst] += h[src]   (64 threads per edge, 4 floats each -> f32 atomics)
// ---------------------------------------------------------------------------
__global__ __launch_bounds__(256) void scatter_add(const float* __restrict__ h,
                                                   const int* __restrict__ srcIdx,
                                                   const int* __restrict__ dstIdx,
                                                   float* __restrict__ z) {
    int t = blockIdx.x * 256 + threadIdx.x;        // 0 .. 19.2M-1
    int edge = t >> 6;
    int c = (t & 63) << 2;
    int s = srcIdx[edge];
    int d = dstIdx[edge];
    float4 v = *(const float4*)(h + s * HID + c);
    float* zp = z + d * HID + c;
    atomicAdd(zp + 0, v.x);
    atomicAdd(zp + 1, v.y);
    atomicAdd(zp + 2, v.z);
    atomicAdd(zp + 3, v.w);
}

// ---------------------------------------------------------------------------
// GEMM1: t = relu(z @ W1 + b1), bf16 output.
// Block = 16 waves, 32-row band. Wave w computes two 16x16 tiles
// (rows 0-15 and 16-31 of the band) at columns [16w,16w+16), sharing B.
// ---------------------------------------------------------------------------
__global__ __launch_bounds__(512) void gemm1_relu(const float* __restrict__ z,
                                                  const __bf16* __restrict__ Wp,
                                                  const float* __restrict__ b1,
                                                  __bf16* __restrict__ tOut) {
    __shared__ __attribute__((aligned(16))) __bf16 zs[32 * HID];   // 16 KB
    int m0 = blockIdx.x * 32;
    int tid = threadIdx.x;

    // Stage 32x256 band of z into LDS as bf16 (coalesced, 16 elems/thread).
    for (int i = tid; i < 32 * HID; i += 512)
        zs[i] = (__bf16)z[(m0 + (i >> 8)) * HID + (i & 255)];
    __syncthreads();

    int wave = tid >> 5, lane = tid & 31;
    int row = lane & 15;
    int kb = (lane >> 4) << 3;                     // 0 or 8 (A-frag K base)
    const __bf16* wBase = Wp + wave * 512 + lane * 16;

    v8f acc0 = {}, acc1 = {};
    #pragma unroll
    for (int kk = 0; kk < HID; kk += 32) {
        BFrag A0, A1, B;
        const uint4* bp = (const uint4*)(wBase + (kk >> 5) * (16 * 512));
        B.q.lo = bp[0];
        B.q.hi = bp[1];
        A0.q.lo = *(const uint4*)&zs[row * HID + kk + kb];
        A0.q.hi = *(const uint4*)&zs[row * HID + kk + 16 + kb];
        A1.q.lo = *(const uint4*)&zs[(row + 16) * HID + kk + kb];
        A1.q.hi = *(const uint4*)&zs[(row + 16) * HID + kk + 16 + kb];
        acc0 = __builtin_amdgcn_wmma_f32_16x16x32_bf16(false, A0.v, false, B.v,
                                                       (short)0, acc0, false, false);
        acc1 = __builtin_amdgcn_wmma_f32_16x16x32_bf16(false, A1.v, false, B.v,
                                                       (short)0, acc1, false, false);
    }

    // Epilogue: bias + relu, store bf16.  C/D layout: VGPR v -> M = v + 8*(lane>>4).
    int col = wave * 16 + (lane & 15);
    float bias = b1[col];
    int rb = m0 + ((lane >> 4) << 3);
    #pragma unroll
    for (int v = 0; v < 8; v++) {
        float v0 = acc0[v] + bias;
        float v1 = acc1[v] + bias;
        v0 = v0 > 0.0f ? v0 : 0.0f;
        v1 = v1 > 0.0f ? v1 : 0.0f;
        tOut[(rb + v) * HID + col] = (__bf16)v0;
        tOut[(rb + 16 + v) * HID + col] = (__bf16)v1;
    }
}

// ---------------------------------------------------------------------------
// GEMM2 + residual + LayerNorm + ReLU fused (32-row band, 2 tiles/wave).
// m = t @ W2 + b2 ; out = relu(LN(m + hIn) * gamma + beta)
// t band staged LDS-direct via GLOBAL_LOAD_ASYNC_TO_LDS_B128 (ASYNCcnt).
// ---------------------------------------------------------------------------
__global__ __launch_bounds__(512) void gemm2_ln(const __bf16* __restrict__ t,
                                                const __bf16* __restrict__ Wp,
                                                const float* __restrict__ b2,
                                                const float* __restrict__ hIn,
                                                const float* __restrict__ gamma,
                                                const float* __restrict__ beta,
                                                float* __restrict__ out) {
    __shared__ __attribute__((aligned(16))) __bf16 ts[32 * HID];   // 16 KB
    __shared__ float ms[32 * HID];                                  // 32 KB
    int m0 = blockIdx.x * 32;
    int tid = threadIdx.x;

#if USE_ASYNC_STAGE
    {
        // 32x256 bf16 band = 16 KB = 1024 x 16B chunks; 2 per thread.
        // VDST = LDS byte address, VADDR = 32-bit byte offset, SADDR = base.
        unsigned ldsBase = (unsigned)(unsigned long long)(const void*)ts;
        const __bf16* gbase = t + m0 * HID;
        unsigned off0 = (unsigned)tid * 16u;
        unsigned off1 = off0 + 8192u;
        asm volatile("global_load_async_to_lds_b128 %0, %1, %2"
                     :: "v"(ldsBase + off0), "v"(off0), "s"(gbase) : "memory");
        asm volatile("global_load_async_to_lds_b128 %0, %1, %2"
                     :: "v"(ldsBase + off1), "v"(off1), "s"(gbase) : "memory");
        asm volatile("s_wait_asynccnt 0" ::: "memory");
    }
#else
    ((uint4*)ts)[tid]       = ((const uint4*)(t + m0 * HID))[tid];
    ((uint4*)ts)[tid + 512] = ((const uint4*)(t + m0 * HID))[tid + 512];
#endif
    __syncthreads();

    int wave = tid >> 5, lane = tid & 31;
    int row = lane & 15;
    int kb = (lane >> 4) << 3;
    const __bf16* wBase = Wp + wave * 512 + lane * 16;

    v8f acc0 = {}, acc1 = {};
    #pragma unroll
    for (int kk = 0; kk < HID; kk += 32) {
        BFrag A0, A1, B;
        const uint4* bp = (const uint4*)(wBase + (kk >> 5) * (16 * 512));
        B.q.lo = bp[0];
        B.q.hi = bp[1];
        A0.q.lo = *(const uint4*)&ts[row * HID + kk + kb];
        A0.q.hi = *(const uint4*)&ts[row * HID + kk + 16 + kb];
        A1.q.lo = *(const uint4*)&ts[(row + 16) * HID + kk + kb];
        A1.q.hi = *(const uint4*)&ts[(row + 16) * HID + kk + 16 + kb];
        acc0 = __builtin_amdgcn_wmma_f32_16x16x32_bf16(false, A0.v, false, B.v,
                                                       (short)0, acc0, false, false);
        acc1 = __builtin_amdgcn_wmma_f32_16x16x32_bf16(false, A1.v, false, B.v,
                                                       (short)0, acc1, false, false);
    }

    // Park m = acc + b2 in LDS for the row-wise LayerNorm.
    int col = wave * 16 + (lane & 15);
    float bias = b2[col];
    int rloc = (lane >> 4) << 3;
    #pragma unroll
    for (int v = 0; v < 8; v++) {
        ms[(rloc + v) * HID + col]      = acc0[v] + bias;
        ms[(rloc + 16 + v) * HID + col] = acc1[v] + bias;
    }
    __syncthreads();

    // LayerNorm: wave w owns rows w and w+16 of the band.
    #pragma unroll
    for (int half = 0; half < 2; half++) {
        int r = wave + half * 16;
        float x[8];
        float sum = 0.0f, sumsq = 0.0f;
        #pragma unroll
        for (int jj = 0; jj < 8; jj++) {
            int c = lane + 32 * jj;
            float xv = ms[r * HID + c] + hIn[(m0 + r) * HID + c];
            x[jj] = xv;
            sum += xv;
            sumsq += xv * xv;
        }
        #pragma unroll
        for (int off = 16; off > 0; off >>= 1) {
            sum   += __shfl_xor(sum,   off, 32);
            sumsq += __shfl_xor(sumsq, off, 32);
        }
        float mu  = sum * (1.0f / HID);
        float var = sumsq * (1.0f / HID) - mu * mu;
        float rstd = rsqrtf(var + LN_EPS);
        #pragma unroll
        for (int jj = 0; jj < 8; jj++) {
            int c = lane + 32 * jj;
            float y = (x[jj] - mu) * rstd * gamma[c] + beta[c];
            y = y > 0.0f ? y : 0.0f;
            out[(m0 + r) * HID + c] = y;
        }
    }
}

// ---------------------------------------------------------------------------
extern "C" void kernel_launch(void* const* d_in, const int* in_sizes, int n_in,
                              void* d_out, int out_size, void* d_ws, size_t ws_size,
                              hipStream_t stream) {
    const float* h     = (const float*)d_in[0];
    const int*   ei    = (const int*)d_in[1];   // [2, N_EDGES]: src then dst
    const float* W1    = (const float*)d_in[2];
    const float* b1    = (const float*)d_in[3];
    const float* W2    = (const float*)d_in[4];
    const float* b2    = (const float*)d_in[5];
    const float* eps   = (const float*)d_in[6];
    const float* gamma = (const float*)d_in[7];
    const float* beta  = (const float*)d_in[8];
    float* out = (float*)d_out;

    char* ws = (char*)d_ws;
    __bf16* Wp   = (__bf16*)ws;                                  // 786,432 B
    float*  zBuf = (float*)(ws + 786432);                        // 102,400,000 B
    __bf16* tBf  = (__bf16*)(ws + 786432 + 102400000);           // 51,200,000 B

    // Pack all 6 weight matrices once (bf16, WMMA B-fragment layout).
    pack_weights<<<1536, 256, 0, stream>>>(W1, W2, Wp);

    for (int l = 0; l < 3; l++) {
        const float* hIn = (l == 0) ? h : out;
        // z = (1+eps)*h
        z_init<<<25000, 256, 0, stream>>>(hIn, eps + l, zBuf);
        // z[dst] += h[src]
        scatter_add<<<75000, 256, 0, stream>>>(hIn, ei, ei + N_EDGES, zBuf);
        // t = relu(z @ W1 + b1)  -> bf16
        gemm1_relu<<<3125, 512, 0, stream>>>(zBuf, Wp + (l * 2 + 0) * 65536,
                                             b1 + l * HID, tBf);
        // out = relu(LN(t @ W2 + b2 + hIn))
        gemm2_ln<<<3125, 512, 0, stream>>>(tBf, Wp + (l * 2 + 1) * 65536,
                                           b2 + l * HID, hIn,
                                           gamma + l * HID, beta + l * HID, out);
    }
}
